// MultiModalGAT_69784628625410
// MI455X (gfx1250) — matrix operand order repaired and checked by reading, not compiled, not run on hardware
//
#include <hip/hip_runtime.h>

// ---------------------------------------------------------------------------
// GATv2 (3 layers, 4 heads x 64) + 2 MLP heads for MI455X (gfx1250, wave32).
// Dense transforms: v_wmma_f32_16x16x32_bf16, fp32 accumulate. Activations are
// converted to bf16 once per layer; GEMM inner loop is pure load+wmma.
// Edge softmax: multi-pass atomic segment reduction (HBM/atomic bound).
// ---------------------------------------------------------------------------

typedef __attribute__((ext_vector_type(16))) __bf16          v16bf;
typedef __attribute__((ext_vector_type(16))) unsigned short  v16us;
typedef __attribute__((ext_vector_type(8)))  unsigned short  v8us;
typedef __attribute__((ext_vector_type(8)))  float           v8f;

#define HEADS 4
#define HID   64
#define TN    256   // HEADS*HID: output width of every GATv2 linear

static __device__ __forceinline__ unsigned short f2bf(float f) {
    unsigned int u = __float_as_uint(f);
    u += 0x7FFFu + ((u >> 16) & 1u);        // round-to-nearest-even
    return (unsigned short)(u >> 16);
}

static __device__ __forceinline__ float atomicMaxFloat(float* addr, float v) {
    if (v >= 0.0f)
        return __int_as_float(atomicMax((int*)addr, __float_as_int(v)));
    else
        return __uint_as_float(atomicMin((unsigned int*)addr, __float_as_uint(v)));
}

// ---------------------------------------------------------------------------
// fp32 -> bf16 bulk convert (8 elements/thread, n % 8 == 0)
// ---------------------------------------------------------------------------
__global__ void k_f32_to_bf16(const float* __restrict__ src,
                              unsigned short* __restrict__ dst, int n) {
    int i = (blockIdx.x * blockDim.x + threadIdx.x) * 8;
    if (i >= n) return;
    const float4* s = (const float4*)(src + i);
    float4 a = s[0], b = s[1];
    v8us r;
    r[0] = f2bf(a.x); r[1] = f2bf(a.y); r[2] = f2bf(a.z); r[3] = f2bf(a.w);
    r[4] = f2bf(b.x); r[5] = f2bf(b.y); r[6] = f2bf(b.z); r[7] = f2bf(b.w);
    *(v8us*)(dst + i) = r;
}

// ---------------------------------------------------------------------------
// Weight transpose + fp32 -> bf16 convert:  W[K,256] -> Wt[256,K]
// ---------------------------------------------------------------------------
__global__ void k_transpose_bf16(const float* __restrict__ W,
                                 unsigned short* __restrict__ Wt, int K) {
    int idx = blockIdx.x * blockDim.x + threadIdx.x;
    if (idx >= K * TN) return;
    int k = idx / TN, n = idx % TN;
    Wt[(size_t)n * K + k] = f2bf(W[idx]);
}

// ---------------------------------------------------------------------------
// out[M,256] = Ab[M,K](bf16) @ W + bias,  W given as bf16 Wt[256,K].
// Block: 256 thr = 8 waves, tile 32(M) x 256(N).
//   wave (w>>2) picks row-tile (16 rows), (w&3) picks 64-col group:
//   4 WMMA accumulators per wave, A fragment reused 4x.
// A frag layout (16x32 bf16): lane l row m0+(l&15); elem i <-> k = 16*(l>>4) + i
// B frag: elem i <-> k = 16*(l>>4) + i, col n0+(l&15) -> contiguous in Wt.
// ---------------------------------------------------------------------------
__global__ __launch_bounds__(256) void k_gemm_bias_wmma(
    const unsigned short* __restrict__ Ab, const unsigned short* __restrict__ Wt,
    const float* __restrict__ bias, float* __restrict__ out, int M, int K)
{
    const int lane = threadIdx.x & 31;
    const int wave = threadIdx.x >> 5;
    const int half = lane >> 4;
    const int l16  = lane & 15;
    const int m0   = blockIdx.x * 32 + (wave >> 2) * 16;
    const int n0   = (wave & 3) * 64;

    int mrow = m0 + l16;
    if (mrow >= M) mrow = M - 1;                 // clamp; pad rows never stored
    const unsigned short* arow = Ab + (size_t)mrow * K + 16 * half;
    const unsigned short* br0  = Wt + (size_t)(n0 + l16) * K + 16 * half;
    const unsigned short* br1  = br0 + (size_t)16 * K;
    const unsigned short* br2  = br0 + (size_t)32 * K;
    const unsigned short* br3  = br0 + (size_t)48 * K;

    v8f c0 = {}, c1 = {}, c2 = {}, c3 = {};

    for (int k0 = 0; k0 < K; k0 += 32) {
        // WGP-scope prefetch of the A stream (shared by the 4 waves of each row-tile)
        __builtin_prefetch(arow + k0 + 128, 0, 3);      // global_prefetch_b8

        v8us alo = *(const v8us*)(arow + k0);           // elems 0..7
        v8us ahi = *(const v8us*)(arow + k0 + 8);       // elems 8..15
        v16us au = __builtin_shufflevector(alo, ahi, 0, 1, 2, 3, 4, 5, 6, 7,
                                           8, 9, 10, 11, 12, 13, 14, 15);
        v16bf a  = __builtin_bit_cast(v16bf, au);

        v16bf b0 = __builtin_bit_cast(v16bf, *(const v16us*)(br0 + k0));
        v16bf b1 = __builtin_bit_cast(v16bf, *(const v16us*)(br1 + k0));
        v16bf b2 = __builtin_bit_cast(v16bf, *(const v16us*)(br2 + k0));
        v16bf b3 = __builtin_bit_cast(v16bf, *(const v16us*)(br3 + k0));

        c0 = __builtin_amdgcn_wmma_f32_16x16x32_bf16(false, a, false, b0,
                                                     (short)0, c0, false, false);
        c1 = __builtin_amdgcn_wmma_f32_16x16x32_bf16(false, a, false, b1,
                                                     (short)0, c1, false, false);
        c2 = __builtin_amdgcn_wmma_f32_16x16x32_bf16(false, a, false, b2,
                                                     (short)0, c2, false, false);
        c3 = __builtin_amdgcn_wmma_f32_16x16x32_bf16(false, a, false, b3,
                                                     (short)0, c3, false, false);
    }

    const float bi0 = bias[n0 + l16];
    const float bi1 = bias[n0 + 16 + l16];
    const float bi2 = bias[n0 + 32 + l16];
    const float bi3 = bias[n0 + 48 + l16];

    if (m0 + 16 <= M) {                          // fast path: unguarded stores
        #pragma unroll
        for (int v = 0; v < 8; ++v) {
            float* o = out + (size_t)(m0 + v + 8 * half) * TN + n0 + l16;
            o[0]  = c0[v] + bi0;
            o[16] = c1[v] + bi1;
            o[32] = c2[v] + bi2;
            o[48] = c3[v] + bi3;
        }
    } else {
        #pragma unroll
        for (int v = 0; v < 8; ++v) {
            int m = m0 + v + 8 * half;
            if (m < M) {
                float* o = out + (size_t)m * TN + n0 + l16;
                o[0]  = c0[v] + bi0;
                o[16] = c1[v] + bi1;
                o[32] = c2[v] + bi2;
                o[48] = c3[v] + bi3;
            }
        }
    }
}

// ---------------------------------------------------------------------------
// Edge phase helpers
// ---------------------------------------------------------------------------
__global__ void k_init_md(float* __restrict__ mx, float* __restrict__ den, int n) {
    int i = blockIdx.x * blockDim.x + threadIdx.x;
    if (i >= n) return;
    mx[i]  = __uint_as_float(0xFF800000u);       // -inf
    den[i] = 0.0f;
}

__global__ void k_zero(float* __restrict__ p, int n) {
    int i = blockIdx.x * blockDim.x + threadIdx.x;
    if (i < n) p[i] = 0.0f;
}

// logits[e,h] = att_h . leaky_relu(xl[src] + xr[dst]);  segment-max into mx
__global__ void k_edge_logits(const float* __restrict__ xl, const float* __restrict__ xr,
                              const int* __restrict__ ei, const float* __restrict__ att,
                              float* __restrict__ elog, float* __restrict__ mx,
                              int E, int NE)
{
    int t = blockIdx.x * blockDim.x + threadIdx.x;
    if (t >= NE * HEADS) return;
    int e = t >> 2, h = t & 3;
    int src, dst;
    if (e < E) { src = ei[e]; dst = ei[E + e]; } else { src = dst = e - E; }

    const float4* pl = (const float4*)(xl + (size_t)src * TN + h * HID);
    const float4* pr = (const float4*)(xr + (size_t)dst * TN + h * HID);
    const float4* pa = (const float4*)(att + h * HID);
    float acc = 0.0f;
    #pragma unroll
    for (int i = 0; i < 16; ++i) {
        float4 a = pl[i], b = pr[i], w = pa[i];
        float sx = a.x + b.x; sx = sx > 0.0f ? sx : 0.2f * sx; acc += sx * w.x;
        float sy = a.y + b.y; sy = sy > 0.0f ? sy : 0.2f * sy; acc += sy * w.y;
        float sz = a.z + b.z; sz = sz > 0.0f ? sz : 0.2f * sz; acc += sz * w.z;
        float sw = a.w + b.w; sw = sw > 0.0f ? sw : 0.2f * sw; acc += sw * w.w;
    }
    elog[t] = acc;
    atomicMaxFloat(mx + (size_t)dst * HEADS + h, acc);
}

// elog[t] = exp(logit - mx[dst,h]);  segment-sum into den
__global__ void k_edge_exp(float* __restrict__ elog, const float* __restrict__ mx,
                           float* __restrict__ den, const int* __restrict__ ei,
                           int E, int NE)
{
    int t = blockIdx.x * blockDim.x + threadIdx.x;
    if (t >= NE * HEADS) return;
    int e = t >> 2, h = t & 3;
    int dst = (e < E) ? ei[E + e] : (e - E);
    float v = __expf(elog[t] - mx[(size_t)dst * HEADS + h]);
    elog[t] = v;
    atomicAdd(den + (size_t)dst * HEADS + h, v);
}

// acc[dst,h,:] += (ex/den) * xl[src,h,:]
__global__ void k_edge_scatter(const float* __restrict__ elog, const float* __restrict__ den,
                               const float* __restrict__ xl, const int* __restrict__ ei,
                               float* __restrict__ acc, int E, int NE)
{
    int t = blockIdx.x * blockDim.x + threadIdx.x;
    if (t >= NE * HEADS) return;
    int e = t >> 2, h = t & 3;
    int src, dst;
    if (e < E) { src = ei[e]; dst = ei[E + e]; } else { src = dst = e - E; }
    float w = elog[t] / den[(size_t)dst * HEADS + h];

    const float4* pl = (const float4*)(xl + (size_t)src * TN + h * HID);
    float* po = acc + (size_t)dst * TN + h * HID;
    #pragma unroll
    for (int i = 0; i < 16; ++i) {
        float4 a = pl[i];
        atomicAdd(po + 4 * i + 0, w * a.x);
        atomicAdd(po + 4 * i + 1, w * a.y);
        atomicAdd(po + 4 * i + 2, w * a.z);
        atomicAdd(po + 4 * i + 3, w * a.w);
    }
}

// concat layers: next-layer bf16 input = elu(acc + bias)
__global__ void k_finalize_elu_bf16(const float* __restrict__ acc,
                                    const float* __restrict__ bias,
                                    unsigned short* __restrict__ hb, int n)
{
    int i = blockIdx.x * blockDim.x + threadIdx.x;
    if (i >= n) return;
    float v = acc[i] + bias[i % TN];
    v = v > 0.0f ? v : (__expf(v) - 1.0f);
    hb[i] = f2bf(v);
}

// last layer: mean over heads + bias (no activation), fp32 out
__global__ void k_finalize_mean(const float* __restrict__ acc, const float* __restrict__ bias,
                                float* __restrict__ h64, int n)
{
    int i = blockIdx.x * blockDim.x + threadIdx.x;
    if (i >= n) return;
    int node = i / HID, c = i % HID;
    const float* r = acc + (size_t)node * TN;
    h64[i] = 0.25f * (r[c] + r[HID + c] + r[2 * HID + c] + r[3 * HID + c]) + bias[c];
}

// ---------------------------------------------------------------------------
// MLP heads: pam50 = relu(h@pw1+pb1)@pw2+pb2 ; surv = relu(h@sw1+sb1)@sw2+sb2
// d_out = [pam50 (N*4) | surv (N)]
// ---------------------------------------------------------------------------
__global__ void k_mlp_heads(const float* __restrict__ h64,
                            const float* __restrict__ pw1, const float* __restrict__ pb1,
                            const float* __restrict__ pw2, const float* __restrict__ pb2,
                            const float* __restrict__ sw1, const float* __restrict__ sb1,
                            const float* __restrict__ sw2, const float* __restrict__ sb2,
                            float* __restrict__ out, int N)
{
    int n = blockIdx.x * blockDim.x + threadIdx.x;
    if (n >= N) return;
    float h[HID];
    const float* hp = h64 + (size_t)n * HID;
    #pragma unroll
    for (int c = 0; c < HID; ++c) h[c] = hp[c];

    float hid[32];
    // pam50 head
    for (int j = 0; j < 32; ++j) {
        float s = pb1[j];
        for (int c = 0; c < HID; ++c) s += h[c] * pw1[c * 32 + j];
        hid[j] = s > 0.0f ? s : 0.0f;
    }
    for (int o = 0; o < 4; ++o) {
        float s = pb2[o];
        for (int j = 0; j < 32; ++j) s += hid[j] * pw2[j * 4 + o];
        out[(size_t)n * 4 + o] = s;
    }
    // survival head
    for (int j = 0; j < 32; ++j) {
        float s = sb1[j];
        for (int c = 0; c < HID; ++c) s += h[c] * sw1[c * 32 + j];
        hid[j] = s > 0.0f ? s : 0.0f;
    }
    float s = sb2[0];
    for (int j = 0; j < 32; ++j) s += hid[j] * sw2[j];
    out[(size_t)N * 4 + n] = s;
}

// ---------------------------------------------------------------------------
extern "C" void kernel_launch(void* const* d_in, const int* in_sizes, int n_in,
                              void* d_out, int out_size, void* d_ws, size_t ws_size,
                              hipStream_t stream)
{
    const float* x  = (const float*)d_in[0];
    const int*   ei = (const int*)d_in[1];
    const int N  = in_sizes[0] / 512;
    const int E  = in_sizes[1] / 2;
    const int NE = E + N;                        // + self loops

    // --- workspace partition (256B aligned) ---
    char* p = (char*)d_ws;
    auto take = [&](size_t bytes) -> char* {
        char* r = p; p += (bytes + 255) & ~(size_t)255; return r;
    };
    unsigned short* Wtl = (unsigned short*)take((size_t)TN * 512 * 2);
    unsigned short* Wtr = (unsigned short*)take((size_t)TN * 512 * 2);
    unsigned short* hb  = (unsigned short*)take((size_t)N * 512 * 2); // bf16 activations
    float* xl   = (float*)take((size_t)N * TN * 4);
    float* xr   = (float*)take((size_t)N * TN * 4);
    float* acc  = xr;                            // xr dead after logits pass
    float* mx   = (float*)take((size_t)N * HEADS * 4);
    float* den  = (float*)take((size_t)N * HEADS * 4);
    float* elog = (float*)take((size_t)NE * HEADS * 4);
    float* h64  = (float*)take((size_t)N * HID * 4);

    struct LayerP { const float *Wl, *bl, *Wr, *br, *att, *bias; int K; };
    LayerP L[3] = {
        { (const float*)d_in[2],  (const float*)d_in[3],  (const float*)d_in[4],
          (const float*)d_in[5],  (const float*)d_in[6],  (const float*)d_in[7],  512 },
        { (const float*)d_in[8],  (const float*)d_in[9],  (const float*)d_in[10],
          (const float*)d_in[11], (const float*)d_in[12], (const float*)d_in[13], 256 },
        { (const float*)d_in[14], (const float*)d_in[15], (const float*)d_in[16],
          (const float*)d_in[17], (const float*)d_in[18], (const float*)d_in[19], 256 },
    };

    const int TB = 256;

    // layer-0 input: x -> bf16 once (shared by both GEMMs of layer 0)
    k_f32_to_bf16<<<((N * 512 / 8) + TB - 1) / TB, TB, 0, stream>>>(x, hb, N * 512);

    for (int l = 0; l < 3; ++l) {
        const int K = L[l].K;
        int wtn = K * TN;
        k_transpose_bf16<<<(wtn + TB - 1) / TB, TB, 0, stream>>>(L[l].Wl, Wtl, K);
        k_transpose_bf16<<<(wtn + TB - 1) / TB, TB, 0, stream>>>(L[l].Wr, Wtr, K);

        int gblocks = (N + 31) / 32;
        k_gemm_bias_wmma<<<gblocks, TB, 0, stream>>>(hb, Wtl, L[l].bl, xl, N, K);
        k_gemm_bias_wmma<<<gblocks, TB, 0, stream>>>(hb, Wtr, L[l].br, xr, N, K);

        k_init_md<<<(N * HEADS + TB - 1) / TB, TB, 0, stream>>>(mx, den, N * HEADS);

        int et = NE * HEADS;
        k_edge_logits<<<(et + TB - 1) / TB, TB, 0, stream>>>(xl, xr, ei, L[l].att,
                                                             elog, mx, E, NE);
        k_zero<<<(N * TN + TB - 1) / TB, TB, 0, stream>>>(acc, N * TN);
        k_edge_exp<<<(et + TB - 1) / TB, TB, 0, stream>>>(elog, mx, den, ei, E, NE);
        k_edge_scatter<<<(et + TB - 1) / TB, TB, 0, stream>>>(elog, den, xl, ei,
                                                              acc, E, NE);
        if (l < 2) {
            // bf16 input for next layer's GEMMs (rounding identical to GEMM-time cvt)
            k_finalize_elu_bf16<<<(N * TN + TB - 1) / TB, TB, 0, stream>>>(
                acc, L[l].bias, hb, N * TN);
        } else {
            k_finalize_mean<<<(N * HID + TB - 1) / TB, TB, 0, stream>>>(
                acc, L[l].bias, h64, N * HID);
        }
    }

    k_mlp_heads<<<(N + TB - 1) / TB, TB, 0, stream>>>(
        h64,
        (const float*)d_in[20], (const float*)d_in[21],
        (const float*)d_in[22], (const float*)d_in[23],
        (const float*)d_in[24], (const float*)d_in[25],
        (const float*)d_in[26], (const float*)d_in[27],
        (float*)d_out, N);
}
